// Net_37598143709627
// MI455X (gfx1250) — matrix operand order repaired and checked by reading, not compiled, not run on hardware
//
#include <hip/hip_runtime.h>
#include <hip/hip_bf16.h>
#include <stdint.h>

// Problem constants (match the reference).
#define NN      100000
#define NE      400000
#define NG      2000
#define IN_DIM  128
#define HID     256
#define OUT_DIM 12

typedef __attribute__((ext_vector_type(16))) _Float16 v16h;
typedef __attribute__((ext_vector_type(8)))  float    v8f;

union Frag16 {
    v16h v;
    uint4 q[2];
};

// ---------------------------------------------------------------- utilities
__global__ void zero_f32_kernel(float* p, long n) {
    long i = (long)blockIdx.x * blockDim.x + threadIdx.x;
    if (i < n) p[i] = 0.0f;
}

// ------------------------------------------------- degree count (per edge)
__global__ void count_kernel(const long long* __restrict__ ei, float* __restrict__ cnt, int E) {
    int e = blockIdx.x * blockDim.x + threadIdx.x;
    if (e >= E) return;
    int d = (int)ei[E + e];
    atomicAdd(&cnt[d], 1.0f);
}

// ---- scatter-add: agg[dst, k] += feat[src, k]; one block per edge, blockDim = D
__global__ void scatter_kernel(const float* __restrict__ feat,
                               const long long* __restrict__ ei,
                               float* __restrict__ agg, int E) {
    int e = blockIdx.x;
    int D = blockDim.x;
    int k = threadIdx.x;
    int s = (int)ei[e];
    int d = (int)ei[E + e];
    atomicAdd(&agg[(long)d * D + k], feat[(long)s * D + k]);
}

// ---- build f16 A matrix: cols [0,Dagg) = agg/max(cnt,1), [Dagg,2*Dagg) = feat
// one block per node, blockDim = 2*Dagg
__global__ void build_a_kernel(const float* __restrict__ agg,
                               const float* __restrict__ cnt,
                               const float* __restrict__ feat,
                               _Float16* __restrict__ abuf) {
    int n    = blockIdx.x;
    int D    = blockDim.x;
    int Dagg = D >> 1;
    int c    = threadIdx.x;
    float v;
    if (c < Dagg) {
        float r = 1.0f / fmaxf(cnt[n], 1.0f);
        v = agg[(long)n * Dagg + c] * r;
    } else {
        v = feat[(long)n * Dagg + (c - Dagg)];
    }
    abuf[(long)n * D + c] = (_Float16)v;
}

// ---- repack W = [Wl; Wr] (row-major [K=2*Khalf, 256]) into WMMA B-fragment
// order: region (kt,nt) -> 32 lanes x 16 halves contiguous per lane.
// lane L holds col n = nt*16 + (L&15), elements j=0..15 -> K = kt*32 + (L>>4)*16 + j
__global__ void pack_w_kernel(const float* __restrict__ Wl,
                              const float* __restrict__ Wr,
                              _Float16* __restrict__ pack, int Khalf) {
    int K = Khalf * 2;
    long i = (long)blockIdx.x * blockDim.x + threadIdx.x;
    long total = (long)K * HID;
    if (i >= total) return;
    int j    = (int)(i & 15);
    int L    = (int)((i >> 4) & 31);
    int reg  = (int)(i >> 9);
    int nt   = reg & 15;       // HID/16 = 16 column tiles
    int kt   = reg >> 4;
    int n    = nt * 16 + (L & 15);
    int kk   = kt * 32 + ((L >> 4) * 16) + j;
    float v  = (kk < Khalf) ? Wl[(long)kk * HID + n]
                            : Wr[(long)(kk - Khalf) * HID + n];
    pack[i] = (_Float16)v;
}

// ---------------------------------------------------------- WMMA f16 GEMM
__device__ __forceinline__ void load_a_frag(Frag16& a, const _Float16* p) {
    a.q[0] = *(const uint4*)(p);
    a.q[1] = *(const uint4*)(p + 8);
}
__device__ __forceinline__ void load_b_frags(Frag16* b, const _Float16* p) {
#pragma unroll
    for (int t = 0; t < 4; ++t) {
        b[t].q[0] = *(const uint4*)(p + t * 512);
        b[t].q[1] = *(const uint4*)(p + t * 512 + 8);
    }
}
__device__ __forceinline__ void wmma4(v8f* acc, const Frag16& a, const Frag16* b) {
#pragma unroll
    for (int t = 0; t < 4; ++t)
        acc[t] = __builtin_amdgcn_wmma_f32_16x16x32_f16(
            false, a.v, false, b[t].v, (short)0, acc[t], false, false);
}

// Out[N, 256] = relu(A[N, K] @ Wcat[K, 256] + bias)
// block = 256 threads = 8 waves; wave (mtile, cgrp): 16 rows x 64 cols.
// K/32 is even (8 or 16) -> unroll-by-2 ping-pong double buffering.
__global__ __launch_bounds__(256)
void gemm_wmma_kernel(const _Float16* __restrict__ A, int K,
                      const _Float16* __restrict__ Wp,
                      const float* __restrict__ bias,
                      float* __restrict__ Out, int N) {
    int lane  = threadIdx.x & 31;
    int wave  = threadIdx.x >> 5;
    int mtile = wave >> 2;          // 0..1
    int cgrp  = wave & 3;           // 0..3 -> 64 cols each
    int m0    = blockIdx.x * 32 + mtile * 16;
    if (m0 >= N) return;

    int row   = m0 + (lane & 15);
    int base0 = (lane >> 4) * 16;   // A layout: 16 contiguous k per lane half

    const _Float16* arow = A + (long)row * K + base0;
    const _Float16* bcol = Wp + ((cgrp * 4) * 32 + lane) * 16;
    // per kt: A advances 32 halves, B advances 16 regions * 512 halves = 8192

    v8f acc[4];
#pragma unroll
    for (int t = 0; t < 4; ++t) acc[t] = (v8f){0.f,0.f,0.f,0.f,0.f,0.f,0.f,0.f};

    Frag16 a0, a1, b0[4], b1[4];
    load_a_frag(a0, arow);
    load_b_frags(b0, bcol);

    int nkt = K >> 5;               // 8 or 16, always even
    for (int kt = 0; kt < nkt; kt += 2) {
        load_a_frag(a1, arow + (kt + 1) * 32);
        load_b_frags(b1, bcol + (long)(kt + 1) * 8192);
        wmma4(acc, a0, b0);
        if (kt + 2 < nkt) {
            load_a_frag(a0, arow + (kt + 2) * 32);
            load_b_frags(b0, bcol + (long)(kt + 2) * 8192);
        }
        wmma4(acc, a1, b1);
    }

    // epilogue: C/D layout -> row = m0 + r + 8*(lane>>4), col = ntile*16 + (lane&15)
    int rbase = 8 * (lane >> 4);
#pragma unroll
    for (int t = 0; t < 4; ++t) {
        int col = cgrp * 64 + t * 16 + (lane & 15);
        float bv = bias[col];
#pragma unroll
        for (int r = 0; r < 8; ++r) {
            float v = acc[t][r] + bv;
            v = fmaxf(v, 0.0f);      // relu
            Out[(long)(m0 + rbase + r) * HID + col] = v;
        }
    }
}

// ----------------------------------------------------- global add pool
// one block per node, blockDim = HID
__global__ void pool_kernel(const float* __restrict__ h,
                            const long long* __restrict__ batch,
                            float* __restrict__ pooled) {
    int n = blockIdx.x;
    int c = threadIdx.x;
    atomicAdd(&pooled[(long)batch[n] * HID + c], h[(long)n * HID + c]);
}

// ------------------------------------------------- head: pooled @ Wlin + blin
__global__ void final_kernel(const float* __restrict__ pooled,
                             const float* __restrict__ Wlin,
                             const float* __restrict__ blin,
                             float* __restrict__ out) {
    int i = blockIdx.x * blockDim.x + threadIdx.x;
    if (i >= NG * OUT_DIM) return;
    int g = i / OUT_DIM;
    int o = i - g * OUT_DIM;
    float s = blin[o];
    const float* p = pooled + (long)g * HID;
#pragma unroll 8
    for (int c = 0; c < HID; ++c) s += p[c] * Wlin[c * OUT_DIM + o];
    out[i] = s;
}

static inline int cdiv(long a, int b) { return (int)((a + b - 1) / b); }

extern "C" void kernel_launch(void* const* d_in, const int* in_sizes, int n_in,
                              void* d_out, int out_size, void* d_ws, size_t ws_size,
                              hipStream_t stream) {
    const float*     x      = (const float*)d_in[0];
    const long long* ei     = (const long long*)d_in[1];
    const long long* batch  = (const long long*)d_in[2];
    const float*     W1l    = (const float*)d_in[3];
    const float*     b1     = (const float*)d_in[4];
    const float*     W1r    = (const float*)d_in[5];
    const float*     W2l    = (const float*)d_in[6];
    const float*     b2     = (const float*)d_in[7];
    const float*     W2r    = (const float*)d_in[8];
    const float*     Wlin   = (const float*)d_in[9];
    const float*     blin   = (const float*)d_in[10];
    float*           out    = (float*)d_out;

    // ---- workspace layout (256B-aligned regions, aliased along the dataflow)
    char* ws = (char*)d_ws;
    size_t off = 0;
    auto alloc = [&](size_t bytes) -> char* {
        char* p = ws + off;
        off += (bytes + 255) & ~(size_t)255;
        return p;
    };
    float*    cnt    = (float*)alloc((size_t)NN * 4);
    float*    pooled = (float*)alloc((size_t)NG * HID * 4);
    _Float16* wp1    = (_Float16*)alloc((size_t)(2 * IN_DIM) * HID * 2);
    _Float16* wp2    = (_Float16*)alloc((size_t)(2 * HID) * HID * 2);
    char*     Ra     = alloc((size_t)NN * HID * 4);      // agg1 / agg2 / h2
    char*     Rb     = alloc((size_t)NN * (2 * HID) * 2);// abuf1 / abuf2
    float*    h1     = (float*)alloc((size_t)NN * HID * 4);

    float*    agg1  = (float*)Ra;                 // [NN,128]
    float*    agg2  = (float*)Ra;                 // [NN,256]
    float*    h2    = (float*)Ra;                 // [NN,256]
    _Float16* abuf1 = (_Float16*)Rb;              // [NN,256]
    _Float16* abuf2 = (_Float16*)Rb;              // [NN,512]

    const int T = 256;

    // ---- layer 1 -----------------------------------------------------------
    zero_f32_kernel<<<cdiv(NN, T), T, 0, stream>>>(cnt, (long)NN);
    zero_f32_kernel<<<cdiv((long)NN * IN_DIM, T), T, 0, stream>>>(agg1, (long)NN * IN_DIM);
    count_kernel<<<cdiv(NE, T), T, 0, stream>>>(ei, cnt, NE);
    scatter_kernel<<<NE, IN_DIM, 0, stream>>>(x, ei, agg1, NE);
    build_a_kernel<<<NN, 2 * IN_DIM, 0, stream>>>(agg1, cnt, x, abuf1);
    pack_w_kernel<<<cdiv((long)(2 * IN_DIM) * HID, T), T, 0, stream>>>(W1l, W1r, wp1, IN_DIM);
    gemm_wmma_kernel<<<cdiv(NN, 32), T, 0, stream>>>(abuf1, 2 * IN_DIM, wp1, b1, h1, NN);

    // ---- layer 2 -----------------------------------------------------------
    zero_f32_kernel<<<cdiv((long)NN * HID, T), T, 0, stream>>>(agg2, (long)NN * HID);
    scatter_kernel<<<NE, HID, 0, stream>>>(h1, ei, agg2, NE);
    build_a_kernel<<<NN, 2 * HID, 0, stream>>>(agg2, cnt, h1, abuf2);
    pack_w_kernel<<<cdiv((long)(2 * HID) * HID, T), T, 0, stream>>>(W2l, W2r, wp2, HID);
    gemm_wmma_kernel<<<cdiv(NN, 32), T, 0, stream>>>(abuf2, 2 * HID, wp2, b2, h2, NN);

    // ---- pool + head -------------------------------------------------------
    zero_f32_kernel<<<cdiv((long)NG * HID, T), T, 0, stream>>>(pooled, (long)NG * HID);
    pool_kernel<<<NN, HID, 0, stream>>>(h2, batch, pooled);
    final_kernel<<<cdiv((long)NG * OUT_DIM, T), T, 0, stream>>>(pooled, Wlin, blin, out);
}
